// EnhancedCondConv2d_39402029973968
// MI455X (gfx1250) — matrix operand order repaired and checked by reading, not compiled, not run on hardware
//
#include <hip/hip_runtime.h>
#include <math.h>

typedef __attribute__((ext_vector_type(16))) _Float16 v16h;
typedef __attribute__((ext_vector_type(8)))  _Float16 v8h;
typedef __attribute__((ext_vector_type(8)))  float    v8f;

#define B_    16
#define CIN_  128
#define COUT_ 256
#define H_    64
#define W_    64
#define E_    4
#define HW_   4096
#define KTOT  1152          // 9 taps * 128 cin  (tap-major K ordering)
#define KSTEPS 36           // KTOT/32 ; 4 k-steps per tap

// ---------------- generic plane mean (used for x-GAP and conv-out pool) ----
__global__ void k_plane_mean(const float* __restrict__ x, float* __restrict__ g) {
  int bc = blockIdx.x;
  const float* p = x + (size_t)bc * HW_;
  float s = 0.f;
  for (int i = threadIdx.x; i < HW_; i += 256) s += p[i];
  __shared__ float red[256];
  red[threadIdx.x] = s; __syncthreads();
  for (int off = 128; off > 0; off >>= 1) {
    if (threadIdx.x < off) red[threadIdx.x] += red[threadIdx.x + off];
    __syncthreads();
  }
  if (threadIdx.x == 0) g[bc] = red[0] * (1.0f / HW_);
}

// ---------------- routing MLP + softmax (tiny) -----------------------------
__global__ void k_route(const float* __restrict__ g,
                        const float* r1w, const float* r1b,
                        const float* r2w, const float* r2b,
                        const float* r3w, const float* r3b,
                        float* __restrict__ rw) {
  int b = threadIdx.x;
  if (b >= B_) return;
  float h1[2];
  for (int j = 0; j < 2; ++j) {
    float s = r1b[j];
    for (int c = 0; c < CIN_; ++c) s += g[b * CIN_ + c] * r1w[j * CIN_ + c];
    h1[j] = fmaxf(s, 0.f);
  }
  float h2[4];
  for (int j = 0; j < 4; ++j) {
    float s = r2b[j];
    for (int i = 0; i < 2; ++i) s += h1[i] * r2w[j * 2 + i];
    h2[j] = fmaxf(s, 0.f);
  }
  float lg[4]; float mx = -1e30f;
  for (int j = 0; j < 4; ++j) {
    float s = r3b[j];
    for (int i = 0; i < 4; ++i) s += h2[i] * r3w[j * 4 + i];
    lg[j] = s; mx = fmaxf(mx, s);
  }
  float den = 0.f;
  for (int j = 0; j < 4; ++j) { lg[j] = __expf(lg[j] - mx); den += lg[j]; }
  for (int j = 0; j < 4; ++j) rw[b * 4 + j] = lg[j] / den;
}

// ---------------- x fp32 -> f16 --------------------------------------------
__global__ void k_cvt(const float* __restrict__ x, _Float16* __restrict__ xh, int n) {
  int i = blockIdx.x * 256 + threadIdx.x;
  if (i < n) xh[i] = (_Float16)x[i];
}

// ---------------- mix expert weights + swizzle into WMMA A-fragment layout -
// Tap-major K: k' = tap*128 + cin, tap = kh*3+kw.
// Fragment storage: [(b*16+mtile)*36 + ks][lane*16 + (v*2+h)], f16.
// A 16x32 f16 lane layout (ISA 7.12.2): lane<16 -> M=lane (lane>=16 same M):
//   VGPR v<4 : kl = (lane/16)*8 + 2v+h ; v>=4 : kl = 16 + (lane/16)*8 + 2(v-4)+h
__global__ void k_wcomb(const float* __restrict__ ew, const float* __restrict__ rw,
                        _Float16* __restrict__ frag) {
  int blk = blockIdx.x;                 // ((b*16 + mt)*36 + ks)
  int ks = blk % KSTEPS;
  int mt = (blk / KSTEPS) % 16;
  int b  = blk / (KSTEPS * 16);
  float w0 = rw[b * 4 + 0], w1 = rw[b * 4 + 1];
  float w2 = rw[b * 4 + 2], w3 = rw[b * 4 + 3];
  const size_t ESTRIDE = (size_t)COUT_ * CIN_ * 9;
  _Float16* o = frag + (size_t)blk * 512;
  for (int idx = threadIdx.x; idx < 512; idx += 256) {
    int lane = idx >> 4, i = idx & 15;
    int v = i >> 1, h = i & 1;
    int khalf = lane >> 4;
    int kl = (v < 4) ? (khalf * 8 + v * 2 + h) : (16 + khalf * 8 + (v - 4) * 2 + h);
    int k = ks * 32 + kl;               // tap-major k'
    int tap = k >> 7;                   // 0..8  (kh*3+kw)
    int cin = k & 127;
    int m = mt * 16 + (lane & 15);
    size_t base = ((size_t)m * CIN_ + cin) * 9 + tap;
    float acc = w0 * ew[base] + w1 * ew[ESTRIDE + base]
              + w2 * ew[2 * ESTRIDE + base] + w3 * ew[3 * ESTRIDE + base];
    o[idx] = (_Float16)acc;
  }
}

// ---------------- main conv: implicit-GEMM via v_wmma_f32_16x16x32_f16 -----
// One block per (b, oh): 8 waves, wave w owns m-tiles {w, w+8}, all 4 n-tiles
// (one full output row).  x rows {oh-1,oh,oh+1} staged once in LDS (48 KB);
// per k-step the 4 shared B fragments are built cooperatively in LDS and
// consumed by all waves as aligned b128 LDS loads -> 8 WMMAs per k-step/wave.
// B 32x16 f16 lane layout: lane -> N=lane&15, kl = (lane>>4)*8 + e (e<8), 16 + (lane>>4)*8 + e-8 (e>=8).
__global__ void __launch_bounds__(256)
k_conv_wmma(const _Float16* __restrict__ xh, const _Float16* __restrict__ frag,
            float* __restrict__ out) {
  __shared__ _Float16 xs[3 * CIN_ * 64];    // [rowIdx][cin][iw]
  __shared__ _Float16 bfrag[4 * 512];       // 4 n-tiles of B fragments

  int tid  = threadIdx.x;
  int lane = tid & 31;
  int wv   = tid >> 5;                      // 0..7
  int oh = blockIdx.x & 63;
  int b  = blockIdx.x >> 6;

  // ---- stage 3 input rows (zero-filled at vertical borders) ----
  const _Float16* xb = xh + (size_t)b * CIN_ * HW_;
  for (int c8 = tid; c8 < 3 * CIN_ * 8; c8 += 256) {   // 16-byte chunks
    int rowIdx = c8 >> 10;                  // /(128*8)
    int rem = c8 & 1023;
    int cin = rem >> 3, p8 = rem & 7;
    int ih = oh + rowIdx - 1;
    uint4 val = make_uint4(0u, 0u, 0u, 0u);
    if ((unsigned)ih < 64u)
      val = *(const uint4*)(xb + ((size_t)cin << 12) + (ih << 6) + p8 * 8);
    *(uint4*)(xs + ((rowIdx * CIN_ + cin) << 6) + p8 * 8) = val;
  }
  __syncthreads();

  const _Float16* A0 = frag + ((size_t)(b * 16 + wv) * KSTEPS) * 512 + lane * 16;
  const _Float16* A1 = A0 + (size_t)8 * KSTEPS * 512;

  // build-phase mapping: tid -> (tile t, fragment lane, half-segment)
  int bt    = tid >> 6;                     // 0..3
  int blane = (tid >> 1) & 31;
  int bi0   = (tid & 1) * 8;
  int bn    = blane & 15;
  int bkl0  = ((blane >> 4) << 3) + bi0 * 2; // kl of first of 8 elements

  v8f acc[2][4] = {};

  for (int ks = 0; ks < KSTEPS; ++ks) {
    int tap = ks >> 2;                      // fixed (kh,kw) per k-step
    int kh  = tap / 3, kw = tap - kh * 3;
    int cin0 = (ks & 3) * 32;

    // ---- cooperative B-fragment build (shared by all 8 waves) ----
    {
      int iw = bt * 16 + bn + kw - 1;
      v8h tmp = (v8h)(_Float16)0.f;
      if ((unsigned)iw < 64u) {
        const _Float16* src = xs + (((kh * CIN_ + cin0 + bkl0) << 6) + iw);
#pragma unroll
        for (int j = 0; j < 8; ++j) tmp[j] = src[j << 6];
      }
      *(v8h*)(bfrag + bt * 512 + blane * 16 + bi0) = tmp;
    }
    __syncthreads();

    // ---- consume: 2 A fragments x 4 B fragments = 8 WMMAs ----
    v16h a0 = *(const v16h*)(A0 + (size_t)ks * 512);
    v16h a1 = *(const v16h*)(A1 + (size_t)ks * 512);
    if (ks + 1 < KSTEPS) {
      __builtin_prefetch(A0 + (size_t)(ks + 1) * 512, 0, 0);
      __builtin_prefetch(A1 + (size_t)(ks + 1) * 512, 0, 0);
    }
#pragma unroll
    for (int t = 0; t < 4; ++t) {
      v16h bf = *(const v16h*)(bfrag + t * 512 + lane * 16);
      acc[0][t] = __builtin_amdgcn_wmma_f32_16x16x32_f16(
          false, a0, false, bf, (short)0, acc[0][t], false, false);
      acc[1][t] = __builtin_amdgcn_wmma_f32_16x16x32_f16(
          false, a1, false, bf, (short)0, acc[1][t], false, false);
    }
    __syncthreads();                        // protect bfrag before next build
  }

  // C/D f32 16x16 layout: VGPR r, lane<16 -> M=r, lane>=16 -> M=8+r, N=lane&15
  int nlo  = lane & 15;
  int mofs = (lane < 16) ? 0 : 8;
#pragma unroll
  for (int mi = 0; mi < 2; ++mi) {
    int mt = wv + mi * 8;
    float* ob = out + ((size_t)(b * COUT_ + mt * 16) * HW_) + oh * 64;
#pragma unroll
    for (int r = 0; r < 8; ++r) {
#pragma unroll
      for (int t = 0; t < 4; ++t)
        ob[(size_t)(mofs + r) * HW_ + t * 16 + nlo] = acc[mi][t][r];
    }
  }
}

// ---------------- SE channel attention -------------------------------------
__global__ void k_ca(const float* __restrict__ p,
                     const float* w1, const float* b1,
                     const float* w2, const float* b2,
                     float* __restrict__ ca) {
  __shared__ float pb[256];
  __shared__ float hid[64];
  int t = threadIdx.x;
  for (int b = 0; b < B_; ++b) {
    pb[t] = p[b * 256 + t];
    __syncthreads();
    if (t < 64) {
      float s = b1[t];
      for (int c = 0; c < 256; ++c) s += pb[c] * w1[t * 256 + c];
      hid[t] = fmaxf(s, 0.f);
    }
    __syncthreads();
    float s = b2[t];
    for (int j = 0; j < 64; ++j) s += hid[j] * w2[t * 64 + j];
    ca[b * 256 + t] = 1.f / (1.f + __expf(-s));
    __syncthreads();
  }
}

// ---------------- 7x7 spatial attention conv + sigmoid ---------------------
__global__ void __launch_bounds__(64)
k_sa(const float* __restrict__ y, const float* __restrict__ sw,
     const float* __restrict__ sb, float* __restrict__ sa) {
  __shared__ float wsm[256 * 49];           // 50 KB of WGP LDS
  int bi = blockIdx.x;                      // b*64 + oh
  int oh = bi & 63, b = bi >> 6;
  for (int i = threadIdx.x; i < 256 * 49; i += 64) wsm[i] = sw[i];
  __syncthreads();
  int ow = threadIdx.x;
  float acc = sb[0];
  const float* yb = y + (size_t)b * COUT_ * HW_;
  for (int c = 0; c < 256; ++c) {
    const float* pl = yb + (size_t)c * HW_;
    const float* wc = wsm + c * 49;
    for (int kh = 0; kh < 7; ++kh) {
      int ih = oh + kh - 3;
      if ((unsigned)ih >= 64u) continue;
      const float* row = pl + ih * 64;
#pragma unroll
      for (int kw = 0; kw < 7; ++kw) {
        int iw = ow + kw - 3;
        if ((unsigned)iw < 64u) acc += row[iw] * wc[kh * 7 + kw];
      }
    }
  }
  sa[bi * 64 + ow] = 1.f / (1.f + __expf(-acc));
}

// ---------------- zero small accumulators ----------------------------------
__global__ void k_zero(float* a, int n) {
  int i = blockIdx.x * 256 + threadIdx.x;
  if (i < n) a[i] = 0.f;
}

// ---------------- modulate by ca*sa, accumulate BN stats -------------------
__global__ void k_mod(float* __restrict__ y, const float* __restrict__ ca,
                      const float* __restrict__ sa,
                      float* __restrict__ bnsum, float* __restrict__ bnsq) {
  int bc = blockIdx.x;                      // b*256 + c
  int c = bc & 255, b = bc >> 8;
  float cav = ca[b * 256 + c];
  const float* sab = sa + (size_t)b * HW_;
  float* pl = y + (size_t)bc * HW_;
  float ls = 0.f, lq = 0.f;
  for (int i = threadIdx.x; i < HW_; i += 256) {
    float v = pl[i] * cav * sab[i];
    pl[i] = v;
    ls += v; lq += v * v;
  }
  __shared__ float rs[256], rq[256];
  rs[threadIdx.x] = ls; rq[threadIdx.x] = lq; __syncthreads();
  for (int off = 128; off > 0; off >>= 1) {
    if (threadIdx.x < off) {
      rs[threadIdx.x] += rs[threadIdx.x + off];
      rq[threadIdx.x] += rq[threadIdx.x + off];
    }
    __syncthreads();
  }
  if (threadIdx.x == 0) {
    atomicAdd(bnsum + c, rs[0]);
    atomicAdd(bnsq + c, rq[0]);
  }
}

// ---------------- BN finalize ----------------------------------------------
__global__ void k_bnfin(const float* bnsum, const float* bnsq,
                        const float* gamma, const float* beta,
                        float* scale, float* shift) {
  int c = threadIdx.x;
  const float n = (float)(B_ * HW_);
  float mean = bnsum[c] / n;
  float var  = bnsq[c] / n - mean * mean;
  float inv  = rsqrtf(var + 1e-5f);
  float sc   = gamma[c] * inv;
  scale[c] = sc;
  shift[c] = beta[c] - mean * sc;
}

// ---------------- normalize + ReLU -----------------------------------------
__global__ void k_final(const float* __restrict__ y, const float* __restrict__ scale,
                        const float* __restrict__ shift, float* __restrict__ out, int n) {
  int i = blockIdx.x * 256 + threadIdx.x;
  if (i >= n) return;
  int c = (i >> 12) & 255;
  out[i] = fmaxf(y[i] * scale[c] + shift[c], 0.f);
}

// ===========================================================================
extern "C" void kernel_launch(void* const* d_in, const int* in_sizes, int n_in,
                              void* d_out, int out_size, void* d_ws, size_t ws_size,
                              hipStream_t stream) {
  (void)in_sizes; (void)n_in; (void)out_size; (void)ws_size;
  const float* x    = (const float*)d_in[0];
  const float* ew   = (const float*)d_in[1];
  const float* r1w  = (const float*)d_in[2];
  const float* r1b  = (const float*)d_in[3];
  const float* r2w  = (const float*)d_in[4];
  const float* r2b  = (const float*)d_in[5];
  const float* r3w  = (const float*)d_in[6];
  const float* r3b  = (const float*)d_in[7];
  const float* ca1w = (const float*)d_in[8];
  const float* ca1b = (const float*)d_in[9];
  const float* ca2w = (const float*)d_in[10];
  const float* ca2b = (const float*)d_in[11];
  const float* saw  = (const float*)d_in[12];
  const float* sab  = (const float*)d_in[13];
  const float* bng  = (const float*)d_in[14];
  const float* bnb  = (const float*)d_in[15];
  float* out = (float*)d_out;

  char* ws = (char*)d_ws;
  size_t off = 0;
  auto carve = [&](size_t bytes) -> char* {
    char* p = ws + off;
    off += (bytes + 255) & ~(size_t)255;
    return p;
  };
  _Float16* xh   = (_Float16*)carve(sizeof(_Float16) * (size_t)B_ * CIN_ * HW_);
  _Float16* frag = (_Float16*)carve(sizeof(_Float16) * (size_t)B_ * 16 * KSTEPS * 512);
  float* y     = (float*)carve(sizeof(float) * (size_t)B_ * COUT_ * HW_);
  float* g     = (float*)carve(sizeof(float) * B_ * CIN_);
  float* rw    = (float*)carve(sizeof(float) * B_ * E_);
  float* pool  = (float*)carve(sizeof(float) * B_ * COUT_);
  float* cav   = (float*)carve(sizeof(float) * B_ * COUT_);
  float* sav   = (float*)carve(sizeof(float) * (size_t)B_ * HW_);
  float* bnsum = (float*)carve(sizeof(float) * 512);   // [sum(256) | sumsq(256)]
  float* bnsq  = bnsum + 256;
  float* scale = (float*)carve(sizeof(float) * 256);
  float* shift = (float*)carve(sizeof(float) * 256);

  const int nx = B_ * CIN_ * HW_;

  k_cvt<<<(nx + 255) / 256, 256, 0, stream>>>(x, xh, nx);
  k_plane_mean<<<B_ * CIN_, 256, 0, stream>>>(x, g);
  k_route<<<1, 32, 0, stream>>>(g, r1w, r1b, r2w, r2b, r3w, r3b, rw);
  k_wcomb<<<B_ * 16 * KSTEPS, 256, 0, stream>>>(ew, rw, frag);
  k_conv_wmma<<<B_ * 64, 256, 0, stream>>>(xh, frag, y);
  k_plane_mean<<<B_ * COUT_, 256, 0, stream>>>(y, pool);
  k_ca<<<1, 256, 0, stream>>>(pool, ca1w, ca1b, ca2w, ca2b, cav);
  k_sa<<<B_ * 64, 64, 0, stream>>>(y, saw, sab, sav);
  k_zero<<<2, 256, 0, stream>>>(bnsum, 512);
  k_mod<<<B_ * COUT_, 256, 0, stream>>>(y, cav, sav, bnsum, bnsq);
  k_bnfin<<<1, 256, 0, stream>>>(bnsum, bnsq, bng, bnb, scale, shift);
  const int nout = B_ * COUT_ * HW_;
  k_final<<<(nout + 255) / 256, 256, 0, stream>>>(y, scale, shift, out, nout);
}
